// HGT_15719580303743
// MI455X (gfx1250) — compile-verified, matrix operand order checked by reading, not run on hardware
//
#include <hip/hip_runtime.h>
#include <math.h>

// ---------------------------------------------------------------------------
// MI455X / gfx1250 HGT forward.
// Dense contractions: v_wmma_f32_16x16x32_bf16 with operands pre-packed into
// bf16 WMMA-fragment-major layout (one v16bf = one contiguous 32B load per
// lane per fragment; 4 WMMAs per K-step per wave via a 16x64 output strip).
// Irregular edge work: f32 atomics, L2-resident (per-phase footprint << 192MB).
// ---------------------------------------------------------------------------

typedef __attribute__((ext_vector_type(16))) __bf16 v16bf;
typedef __attribute__((ext_vector_type(8)))  float  v8f;

#define DEVI __device__ __forceinline__

DEVI float gelu_f(float x) {
  return 0.5f * x * (1.0f + erff(x * 0.70710678118654752440f));
}

DEVI void atomicMaxF(float* addr, float v) {
  // Monotonic int mapping: non-negative floats compare as ints,
  // negative floats compare reversed as uints. Init value must be -inf.
  if (v >= 0.0f) atomicMax((int*)addr, __float_as_int(v));
  else           atomicMin((unsigned int*)addr, __float_as_uint(v));
}

// ------------------------------- fill --------------------------------------
__global__ void k_fill(float* __restrict__ p, float v, long n) {
  long i = (long)blockIdx.x * blockDim.x + threadIdx.x;
  if (i < n) p[i] = v;
}

// --------------------- fp32 -> bf16 fragment packing -----------------------
// Fragment layout (per CDNA5 ISA 16-bit A 16x32): element index within the
// v16bf of lane L: row m = L&15, K-subchunk ks = (L>=16)*8,
// elems 0..7 -> K = kc*32+ks+0..7, elems 8..15 -> K = kc*32+16+ks+0..7.
// Packed address: (((tile*KC + kc)*32 + lane)*16 + elem), tile over M/16 (A)
// or N/16 (W). Each thread converts one contiguous K-pair -> one u32 store.

__global__ void k_packA(const float* __restrict__ A, __bf16* __restrict__ out,
                        int M, int K) {
  int KC = K >> 5;
  long idx = (long)blockIdx.x * blockDim.x + threadIdx.x;
  long tot = (long)M * K / 2;
  if (idx >= tot) return;
  int  j    = (int)(idx & 7);
  int  lane = (int)((idx >> 3) & 31);
  long rest = idx >> 8;
  int  kc   = (int)(rest % KC);
  int  tile = (int)(rest / KC);
  int  row  = (tile << 4) + (lane & 15);
  int  half = lane >> 4;
  int  e0   = 2 * j;
  int  koff = (e0 < 8) ? e0 : e0 + 8;
  long src  = (long)row * K + (kc << 5) + (half << 3) + koff;
  union { __bf16 h[2]; unsigned u; } p;
  p.h[0] = (__bf16)A[src];
  p.h[1] = (__bf16)A[src + 1];
  ((unsigned*)out)[idx] = p.u;
}

__global__ void k_packW(const float* __restrict__ W, __bf16* __restrict__ out,
                        int K, int N) {
  int KC = K >> 5;
  long idx = (long)blockIdx.x * blockDim.x + threadIdx.x;
  long tot = (long)K * N / 2;
  if (idx >= tot) return;
  int  j    = (int)(idx & 7);
  int  lane = (int)((idx >> 3) & 31);
  long rest = idx >> 8;
  int  kc   = (int)(rest % KC);
  int  tile = (int)(rest / KC);
  int  col  = (tile << 4) + (lane & 15);
  int  half = lane >> 4;
  int  e0   = 2 * j;
  int  koff = (e0 < 8) ? e0 : e0 + 8;
  int  k    = (kc << 5) + (half << 3) + koff;
  union { __bf16 h[2]; unsigned u; } p;
  p.h[0] = (__bf16)W[(long)k * N + col];
  p.h[1] = (__bf16)W[(long)(k + 1) * N + col];
  ((unsigned*)out)[idx] = p.u;
}

// ---------------------- WMMA GEMM on packed fragments ----------------------
// C[M,N] = act(A@W + bias). One wave computes a 16 x (16*TPW) strip: the A
// fragment is loaded once per K-step and reused across TPW WMMAs.
template <int TPW>
__global__ void k_gemm_frag(const __bf16* __restrict__ Af, const __bf16* __restrict__ Wf,
                            const float* __restrict__ bias, float* __restrict__ C,
                            int M, int N, int K, int act)
{
  int wave = (int)(((long)blockIdx.x * blockDim.x + threadIdx.x) >> 5);
  int lane = threadIdx.x & 31;
  int KC = K >> 5;
  int strips = N / (16 * TPW);
  int tMi = wave / strips;
  int sNi = wave % strips;
  if (tMi * 16 >= M) return;           // wave-uniform: EXEC stays all-ones
  const v16bf* Av = (const v16bf*)Af;
  const v16bf* Wv = (const v16bf*)Wf;
  v8f acc[TPW] = {};
  for (int kc = 0; kc < KC; ++kc) {
    v16bf a = Av[((long)tMi * KC + kc) * 32 + lane];
#pragma unroll
    for (int j = 0; j < TPW; ++j) {
      v16bf b = Wv[((long)(sNi * TPW + j) * KC + kc) * 32 + lane];
      acc[j] = __builtin_amdgcn_wmma_f32_16x16x32_bf16(
          false, a, false, b, (short)0, acc[j], false, false);
    }
  }
  int half = lane >> 4, mn = lane & 15;
#pragma unroll
  for (int j = 0; j < TPW; ++j) {
    int col = (sNi * TPW + j) * 16 + mn;
    float bv = bias ? bias[col] : 0.0f;
#pragma unroll
    for (int r = 0; r < 8; ++r) {
      int row = tMi * 16 + r + half * 8;  // C/D layout: lanes>=16 hold M=r+8
      float v = acc[j][r] + bv;
      if (act) v = gelu_f(v);
      C[(long)row * N + col] = v;
    }
  }
}

// ------------------- relation transform: nhd,hde -> nhe --------------------
__global__ void k_rel(const float* __restrict__ kqv, int ld, int off,
                      const float* __restrict__ R, float* __restrict__ out,
                      int n, int H, int D)
{
  long i = (long)blockIdx.x * blockDim.x + threadIdx.x;
  long tot = (long)n * H * D;
  if (i >= tot) return;
  int  e  = (int)(i % D);
  int  h  = (int)((i / D) % H);
  long nd = i / ((long)D * H);
  const float* kp = kqv + nd * ld + off + (long)h * D;
  const float* rp = R + (long)h * D * D + e;
  float s = 0.f;
  for (int d = 0; d < D; ++d) s += kp[d] * rp[(long)d * D];
  out[nd * (long)(H * D) + (long)h * D + e] = s;
}

// -------------------- per-edge attention score + seg-max -------------------
__global__ void k_edge_score(const int* __restrict__ src, const int* __restrict__ dst,
                             int E, const float* __restrict__ krel,
                             const float* __restrict__ kqv, int ld, int qoff,
                             const float* __restrict__ prel, int H, int D,
                             float* __restrict__ score, float* __restrict__ amax)
{
  long i = (long)blockIdx.x * blockDim.x + threadIdx.x;
  if (i >= (long)E * H) return;
  int  h = (int)(i % H);
  long e = i / H;
  int  s = src[e], d = dst[e];
  const float* kp = krel + (long)s * (H * D) + (long)h * D;
  const float* qp = kqv  + (long)d * ld + qoff + (long)h * D;
  float a = 0.f;
  for (int c = 0; c < D; ++c) a += kp[c] * qp[c];
  a *= prel[h] * rsqrtf((float)D);
  score[e * H + h] = a;
  atomicMaxF(&amax[(long)d * H + h], a);
}

// ------------------------- exp(a - max) + seg-sum --------------------------
__global__ void k_edge_expsum(const int* __restrict__ dst, int E,
                              const float* __restrict__ score,
                              const float* __restrict__ amax,
                              float* __restrict__ ex, float* __restrict__ den, int H)
{
  long i = (long)blockIdx.x * blockDim.x + threadIdx.x;
  if (i >= (long)E * H) return;
  int  h = (int)(i % H);
  long e = i / H;
  int  d = dst[e];
  float m = amax[(long)d * H + h];
  if (!isfinite(m)) m = 0.f;
  float v = expf(score[e * H + h] - m);
  ex[e * H + h] = v;
  atomicAdd(&den[(long)d * H + h], v);
}

// -------------------------- message scatter-sum ----------------------------
__global__ void k_edge_scatter(const int* __restrict__ src, const int* __restrict__ dst,
                               int E, const float* __restrict__ ex,
                               const float* __restrict__ den,
                               const float* __restrict__ vrel,
                               float* __restrict__ agg, int H, int D)
{
  int  CO = H * D;
  long i = (long)blockIdx.x * blockDim.x + threadIdx.x;
  if (i >= (long)E * CO) return;
  int  c = (int)(i % CO);
  long e = i / CO;
  int  h = c / D;
  int  s = src[e], d = dst[e];
  float alpha = ex[e * H + h] / (den[(long)d * H + h] + 1e-16f);
  atomicAdd(&agg[(long)d * CO + c], vrel[(long)s * CO + c] * alpha);
}

// ------------------------------- elementwise -------------------------------
__global__ void k_gelu(const float* __restrict__ in, float* __restrict__ out, long n) {
  long i = (long)blockIdx.x * blockDim.x + threadIdx.x;
  if (i < n) out[i] = gelu_f(in[i]);
}

__global__ void k_skip(const float* __restrict__ sk, float* __restrict__ o,
                       const float* __restrict__ x, long n) {
  long i = (long)blockIdx.x * blockDim.x + threadIdx.x;
  if (i >= n) return;
  float s = 1.f / (1.f + expf(-sk[0]));
  o[i] = s * o[i] + (1.f - s) * x[i];
}

// --------------------------------- pooling ---------------------------------
__global__ void k_pool_edge(const int* __restrict__ src, const int* __restrict__ dst,
                            int E, const float* __restrict__ x, float* __restrict__ agg) {
  long i = (long)blockIdx.x * blockDim.x + threadIdx.x;
  if (i >= (long)E * 32) return;
  int  c = (int)(i & 31);
  long e = i >> 5;
  atomicAdd(&agg[(long)dst[e] * 32 + c], x[(long)src[e] * 32 + c]);
}

__global__ void k_pool_score(const float* __restrict__ x, const float* __restrict__ pagg,
                             const float* __restrict__ w_rel, const float* __restrict__ b_rel,
                             const float* __restrict__ w_root, float* __restrict__ out, int n) {
  int i = blockIdx.x * blockDim.x + threadIdx.x;
  if (i >= n) return;
  float s = b_rel[0];
  for (int c = 0; c < 32; ++c)
    s += pagg[(long)i * 32 + c] * w_rel[c] + x[(long)i * 32 + c] * w_root[c];
  out[i] = tanhf(s);
}

// Per group (2048 nodes): iterated argmax -> top-16 (sorted desc, lowest-index
// tiebreak matching lax.top_k), then write scaled 16x32 feature block.
__global__ void k_topk(const float* __restrict__ s, const float* __restrict__ x,
                       float* __restrict__ xsel) {
  __shared__ float sv[2048];
  __shared__ float rv[256];
  __shared__ int   ri[256];
  __shared__ float selv[16];
  __shared__ int   seli[16];
  int g = blockIdx.x, t = threadIdx.x;
  for (int i = t; i < 2048; i += 256) sv[i] = s[(long)g * 2048 + i];
  __syncthreads();
  for (int kk = 0; kk < 16; ++kk) {
    float bv = -INFINITY; int bi = 0x7fffffff;
    for (int i = t; i < 2048; i += 256) {
      float v = sv[i];
      if (v > bv || (v == bv && i < bi)) { bv = v; bi = i; }
    }
    rv[t] = bv; ri[t] = bi;
    __syncthreads();
    for (int off = 128; off > 0; off >>= 1) {
      if (t < off) {
        float ov = rv[t + off]; int oi = ri[t + off];
        if (ov > rv[t] || (ov == rv[t] && oi < ri[t])) { rv[t] = ov; ri[t] = oi; }
      }
      __syncthreads();
    }
    if (t == 0) { selv[kk] = rv[0]; seli[kk] = ri[0]; sv[ri[0]] = -INFINITY; }
    __syncthreads();
  }
  for (int i = t; i < 512; i += 256) {
    int j = i >> 5, c = i & 31;
    long node = (long)g * 2048 + seli[j];
    xsel[(long)g * 512 + i] = x[node * 32 + c] * selv[j];
  }
}

// ------------------------------ tiny MLP (M=8) -----------------------------
__global__ void k_mlp(const float* __restrict__ in, const float* __restrict__ w,
                      const float* __restrict__ b, float* __restrict__ out,
                      int M, int N, int K, int act) {
  int i = blockIdx.x * blockDim.x + threadIdx.x;
  if (i >= M * N) return;
  int r = i / N, c = i % N;
  float s = b[c];
  for (int k = 0; k < K; ++k) s += in[(long)r * K + k] * w[(long)k * N + c];
  out[i] = act ? gelu_f(s) : s;
}

// ===========================================================================
// Host orchestration
// ===========================================================================
//
// Input flattening (JAX pytree, dict keys sorted alphabetically):
//   params = {conv, mlp, pool, proj}
//   conv[0] (idx 0..18):
//     0 kqv.base.b(1536) 1 kqv.base.w(512x1536) 2 kqv.instr.b 3 kqv.instr.w
//     4 out.base.b 5 out.base.w 6 out.instr.b(512) 7 out.instr.w(512x512)
//     8 rel.base_rel.k(8,64,64) 9 .p(8) 10 .v  11-13 rel.ctrl.{k,p,v}
//     14-16 rel.data.{k,p,v}  17 skip.base 18 skip.instr
//   conv[L], L=1..4 (base 19+(L-1)*11):
//     +0 kqv.instr.b +1 kqv.instr.w +2 out.instr.b +3 out.instr.w
//     +4..6 rel.ctrl.{k,p,v} +7..9 rel.data.{k,p,v} +10 skip.instr
//   mlp[i] (63+2i): b, w        (8 layers -> 63..78)
//   pool: 79 b_rel 80 w_rel(32) 81 w_root(32)
//   proj[i] (82+4i): base.b, base.w, instr.b, instr.w  (82..93)
//   data: 94 x_base 95 x_instr 96 ei_data(2,131072) 97 ei_ctrl 98 ei_base(2,16384)
//         99 instr_batch (unused)

extern "C" void kernel_launch(void* const* d_in, const int* in_sizes, int n_in,
                              void* d_out, int out_size, void* d_ws, size_t ws_size,
                              hipStream_t stream)
{
  (void)in_sizes; (void)n_in; (void)out_size; (void)ws_size;

  const int NI = 16384, NB = 4096, EII = 131072, EBI = 16384;
  const int IX_XBASE = 94, IX_XINSTR = 95, IX_EDATA = 96, IX_ECTRL = 97, IX_EBASE = 98;
  (void)IX_XBASE;

  auto P  = [&](int i) -> const float* { return (const float*)d_in[i]; };
  auto PI = [&](int i) -> const int*   { return (const int*)d_in[i]; };

  // bump allocator over d_ws (256B aligned)
  char* wsb = (char*)d_ws;
  size_t off = 0;
  auto alloc = [&](long nfloats) -> float* {
    float* p = (float*)(wsb + off);
    off += (((size_t)nfloats * 4 + 255) & ~(size_t)255);
    return p;
  };

  float* X0    = alloc((long)NI * 512);
  float* X1    = alloc((long)NI * 512);
  float* XB0   = alloc((long)NB * 512);
  float* XB1   = alloc((long)NB * 512);
  float* KQVI  = alloc((long)NI * 1536);
  float* KQVB  = alloc((long)NB * 1536);
  float* KREL  = alloc((long)NI * 512);     // also reused as gelu(agg)
  float* VRELD = alloc((long)NI * 512);
  float* VRELC = alloc((long)NI * 512);
  float* VRELB = alloc((long)NB * 512);
  const long ETOT = 2L * EII + EBI;
  float* SCORE = alloc(ETOT * 8);
  float* EX    = alloc(ETOT * 8);
  float* AMAX  = alloc((long)NI * 8);
  float* DEN   = alloc((long)NI * 8);
  float* AGG   = alloc((long)NI * 512);
  float* PAGG  = alloc((long)NI * 32);
  float* SCP   = alloc(NI);
  float* XSEL  = alloc(8 * 512);
  float* MLPA  = alloc(8 * 512);
  float* MLPB  = alloc(8 * 512);
  __bf16* APACK = (__bf16*)alloc((long)NI * 256);   // 16384x512 bf16 max
  __bf16* WPACK = (__bf16*)alloc(512L * 1536 / 2);  // 512x1536 bf16 max

  auto blocks = [](long n) { return (int)((n + 255) / 256); };
  auto fill = [&](float* p, float v, long n) {
    k_fill<<<blocks(n), 256, 0, stream>>>(p, v, n);
  };
  // pack A/W to bf16 fragments, then strip-mined WMMA GEMM (stream-ordered,
  // so the single APACK/WPACK staging buffers are safely reused per call)
  auto gemm = [&](const float* A, const float* W, const float* b, float* C,
                  int M, int N, int K, int act) {
    k_packA<<<blocks((long)M * K / 2), 256, 0, stream>>>(A, APACK, M, K);
    k_packW<<<blocks((long)K * N / 2), 256, 0, stream>>>(W, WPACK, K, N);
    if (N % 64 == 0) {
      long thr = (long)(M / 16) * (N / 64) * 32;
      k_gemm_frag<4><<<blocks(thr), 256, 0, stream>>>(APACK, WPACK, b, C, M, N, K, act);
    } else {                                    // N = 96 (3co of conv4)
      long thr = (long)(M / 16) * (N / 32) * 32;
      k_gemm_frag<2><<<blocks(thr), 256, 0, stream>>>(APACK, WPACK, b, C, M, N, K, act);
    }
  };
  auto rel = [&](const float* kqv, int ld, int o, const float* R, float* out,
                 int n, int H, int D) {
    k_rel<<<blocks((long)n * H * D), 256, 0, stream>>>(kqv, ld, o, R, out, n, H, D);
  };
  auto escore = [&](const int* src, const int* dst, int E, const float* krel,
                    const float* kqv, int ld, int qoff, const float* p,
                    int H, int D, float* sc) {
    k_edge_score<<<blocks((long)E * H), 256, 0, stream>>>(src, dst, E, krel, kqv,
                                                          ld, qoff, p, H, D, sc, AMAX);
  };
  auto expsum = [&](const int* dst, int E, const float* sc, float* ex, int H) {
    k_edge_expsum<<<blocks((long)E * H), 256, 0, stream>>>(dst, E, sc, AMAX, ex, DEN, H);
  };
  auto scat = [&](const int* src, const int* dst, int E, const float* ex,
                  const float* vrel, int H, int D) {
    k_edge_scatter<<<blocks((long)E * H * D), 256, 0, stream>>>(src, dst, E, ex,
                                                                DEN, vrel, AGG, H, D);
  };

  const int* ed = PI(IX_EDATA);   // src = ed, dst = ed+EII
  const int* ec = PI(IX_ECTRL);
  const int* eb = PI(IX_EBASE);   // src = eb, dst = eb+EBI

  // ----------------------------- projections ------------------------------
  const int PF[4] = {64, 128, 256, 512};
  const float* ci_ = P(IX_XINSTR);
  const float* cb_ = P(94);
  float* idst[3] = {X0, X1, X0};
  float* bdst[3] = {XB0, XB1, XB0};
  for (int i = 0; i < 3; ++i) {
    gemm(cb_, P(82 + 4 * i + 1), P(82 + 4 * i + 0), bdst[i], NB, PF[i + 1], PF[i], 1); // gelu
    gemm(ci_, P(82 + 4 * i + 3), P(82 + 4 * i + 2), idst[i], NI, PF[i + 1], PF[i], 0);
    cb_ = bdst[i]; ci_ = idst[i];
  }

  // ------------------------------ HGT convs -------------------------------
  const int CI[5] = {512, 512, 256, 128, 64};
  const int CO[5] = {512, 256, 128, 64, 32};
  const int HH[5] = {8, 8, 8, 4, 4};
  const float* cur = X0;  float* nxt = X1;
  for (int L = 0; L < 5; ++L) {
    int ci = CI[L], co = CO[L], H = HH[L], D = co / H, tc = 3 * co;
    int kqv_b, kqv_w, out_b, out_w, ck, cp, cv, dk, dp, dv;
    if (L == 0) { kqv_b = 2; kqv_w = 3; out_b = 6; out_w = 7;
                  ck = 11; cp = 12; cv = 13; dk = 14; dp = 15; dv = 16; }
    else { int b0 = 19 + (L - 1) * 11;
           kqv_b = b0; kqv_w = b0 + 1; out_b = b0 + 2; out_w = b0 + 3;
           ck = b0 + 4; cp = b0 + 5; cv = b0 + 6; dk = b0 + 7; dp = b0 + 8; dv = b0 + 9; }

    gemm(cur, P(kqv_w), P(kqv_b), KQVI, NI, tc, ci, 0);
    if (L == 0) gemm(XB0, P(1), P(0), KQVB, NB, tc, ci, 0);

    fill(AMAX, -INFINITY, (long)NI * H);
    fill(DEN, 0.f, (long)NI * H);
    fill(AGG, 0.f, (long)NI * co);

    // relation: data (instr->instr), score block offset 0
    rel(KQVI, tc, 0,      P(dk), KREL,  NI, H, D);
    rel(KQVI, tc, 2 * co, P(dv), VRELD, NI, H, D);
    escore(ed, ed + EII, EII, KREL, KQVI, tc, co, P(dp), H, D, SCORE);
    // relation: ctrl, offset EII*H
    rel(KQVI, tc, 0,      P(ck), KREL,  NI, H, D);
    rel(KQVI, tc, 2 * co, P(cv), VRELC, NI, H, D);
    escore(ec, ec + EII, EII, KREL, KQVI, tc, co, P(cp), H, D, SCORE + (long)EII * H);
    // relation: base_rel (base->instr), layer 0 only, offset 2*EII*H
    if (L == 0) {
      rel(KQVB, tc, 0,      P(8),  KREL,  NB, H, D);
      rel(KQVB, tc, 2 * co, P(10), VRELB, NB, H, D);
      escore(eb, eb + EBI, EBI, KREL, KQVI, tc, co, P(9), H, D, SCORE + 2L * EII * H);
    }

    expsum(ed + EII, EII, SCORE, EX, H);
    expsum(ec + EII, EII, SCORE + (long)EII * H, EX + (long)EII * H, H);
    if (L == 0) expsum(eb + EBI, EBI, SCORE + 2L * EII * H, EX + 2L * EII * H, H);

    scat(ed, ed + EII, EII, EX, VRELD, H, D);
    scat(ec, ec + EII, EII, EX + (long)EII * H, VRELC, H, D);
    if (L == 0) scat(eb, eb + EBI, EBI, EX + 2L * EII * H, VRELB, H, D);

    // out projection: out_lin(gelu(agg)) (+ gated skip at L==0, in_ch==out_ch)
    k_gelu<<<blocks((long)NI * co), 256, 0, stream>>>(AGG, KREL, (long)NI * co);
    gemm(KREL, P(out_w), P(out_b), nxt, NI, co, co, 0);
    if (L == 0) k_skip<<<blocks((long)NI * co), 256, 0, stream>>>(P(18), nxt, cur,
                                                                  (long)NI * co);
    const float* t = cur; cur = nxt; nxt = (float*)t;
  }
  // cur now holds final instr features (16384 x 32)

  // ------------------------------- pooling --------------------------------
  fill(PAGG, 0.f, (long)NI * 32);
  k_pool_edge<<<blocks((long)EII * 32), 256, 0, stream>>>(ed, ed + EII, EII, cur, PAGG);
  k_pool_edge<<<blocks((long)EII * 32), 256, 0, stream>>>(ec, ec + EII, EII, cur, PAGG);
  k_pool_score<<<blocks(NI), 256, 0, stream>>>(cur, PAGG, P(80), P(79), P(81), SCP, NI);
  k_topk<<<8, 256, 0, stream>>>(SCP, cur, XSEL);

  // --------------------------------- MLP ----------------------------------
  const int MD[9] = {512, 256, 128, 64, 32, 16, 8, 4, 1};
  const float* min_ = XSEL;
  float* mbuf[2] = {MLPA, MLPB};
  for (int i = 0; i < 8; ++i) {
    float* o = (i == 7) ? (float*)d_out : mbuf[i & 1];
    k_mlp<<<blocks(8L * MD[i + 1]), 256, 0, stream>>>(min_, P(63 + 2 * i + 1),
                                                      P(63 + 2 * i), o,
                                                      8, MD[i + 1], MD[i], i < 7);
    min_ = o;
  }
}
